// VQVAEBase_60215441490428
// MI455X (gfx1250) — matrix operand order repaired and checked by reading, not compiled, not run on hardware
//
#include <hip/hip_runtime.h>

// Problem constants (fixed by the reference)
#define BB   32
#define DD   64
#define HH   64
#define WW   64
#define NN   (HH * WW)     // 4096 positions per batch image
#define KK   512           // codebook entries
#define WGPOS 128          // positions per workgroup
#define NTHREADS 256       // 8 wave32s
#define EPAD 72            // padded row stride (halves) for enc tile in LDS

typedef __attribute__((ext_vector_type(16))) _Float16 v16h;
typedef __attribute__((ext_vector_type(8)))  _Float16 v8h;
typedef __attribute__((ext_vector_type(4)))  _Float16 v4h;
typedef __attribute__((ext_vector_type(8)))  float    v8f;

union V16 { v16h v; v8h h[2]; };

__global__ __launch_bounds__(NTHREADS, 1)
void vq_argmin_gather_kernel(const float* __restrict__ enc,
                             const float* __restrict__ weight,
                             int* __restrict__ idx_out,
                             float* __restrict__ emb_out)
{
    // LDS: -2*codebook as f16 (64KB), ||w||^2 f32 (2KB), transposed enc tile f16 (18KB)
    __shared__ _Float16 lds_w[KK * DD];
    __shared__ float    lds_wsq[KK];
    __shared__ _Float16 lds_e[WGPOS * EPAD];

    const int tid  = threadIdx.x;
    const int wg   = blockIdx.x;           // 0..1023
    const int b    = wg >> 5;              // batch image (NN/WGPOS == 32 tiles per image)
    const int nbase = (wg & 31) * WGPOS;   // first position of this block's tile

    // ---- Stage codebook as -2*w in f16 (vectorized: float4 -> 4 halves) ----
    for (int i = tid * 4; i < KK * DD; i += NTHREADS * 4) {
        float4 wv = *(const float4*)(weight + i);
        v4h p;
        p[0] = (_Float16)(-2.0f * wv.x);
        p[1] = (_Float16)(-2.0f * wv.y);
        p[2] = (_Float16)(-2.0f * wv.z);
        p[3] = (_Float16)(-2.0f * wv.w);
        *(v4h*)(lds_w + i) = p;
    }

    // ---- ||w_k||^2 in exact f32 (codebook is L2-hot; 2 rows per thread) ----
    for (int k = tid; k < KK; k += NTHREADS) {
        const float* wr = weight + (size_t)k * DD;
        float s = 0.0f;
        #pragma unroll
        for (int d = 0; d < DD; ++d) s += wr[d] * wr[d];
        lds_wsq[k] = s;
    }

    // ---- Stage enc tile transposed: lds_e[i][d] = enc[b, d, nbase+i] (f16) ----
    // Global reads coalesced along n (contiguous axis).
    for (int t = tid; t < DD * WGPOS; t += NTHREADS) {
        const int d = t >> 7;       // /128
        const int i = t & 127;
        float v = enc[(size_t)b * DD * NN + (size_t)d * NN + (nbase + i)];
        lds_e[i * EPAD + d] = (_Float16)v;
    }
    __syncthreads();

    // ---- Per-wave WMMA distance + argmin over all 512 codes ----
    const int wave = tid >> 5;
    const int lane = tid & 31;
    const int half = lane >> 4;     // 0: lanes 0-15, 1: lanes 16-31
    const int lm   = lane & 15;

    // B-fragments (positions): column = lane%16 = position, K(d) split by lane half.
    // kstep s, half h -> d halves at {s*32 + h*8 + [0..7], s*32 + 16 + h*8 + [0..7]}. Loaded once, reused for all 32 code tiles.
    const _Float16* erow = &lds_e[(wave * 16 + lm) * EPAD];
    V16 bf0, bf1;
    bf0.h[0] = *(const v8h*)(erow + half * 8);
    bf0.h[1] = *(const v8h*)(erow + 16 + half * 8);
    bf1.h[0] = *(const v8h*)(erow + 32 + half * 8);
    bf1.h[1] = *(const v8h*)(erow + 48 + half * 8);

    float minval = 3.0e38f;
    int   minidx = 0;

    for (int tile = 0; tile < 32; ++tile) {
        // C init = ||w||^2 for this lane's 8 rows (codes tile*16 + half*8 + r)
        const float* wsqp = &lds_wsq[tile * 16 + half * 8];
        v8f acc;
        #pragma unroll
        for (int r = 0; r < 8; ++r) acc[r] = wsqp[r];

        // A-fragments (codes): row M = lane%16 -> code tile*16+lm.
        // kstep s, half h -> d halves at {s*32 + h*8 + [0..7], s*32 + 16 + h*8 + [0..7]}
        const _Float16* wrow = &lds_w[(tile * 16 + lm) * DD + half * 8];
        V16 a0, a1;
        a0.h[0] = *(const v8h*)(wrow);
        a0.h[1] = *(const v8h*)(wrow + 16);
        a1.h[0] = *(const v8h*)(wrow + 32);
        a1.h[1] = *(const v8h*)(wrow + 48);

        // acc = (-2W)*E + ||w||^2  over d=0..63  ==  squared distance (minus enc_sq const)
        acc = __builtin_amdgcn_wmma_f32_16x16x32_f16(false, a0.v, false, bf0.v,
                                                     (short)0, acc, false, false);
        acc = __builtin_amdgcn_wmma_f32_16x16x32_f16(false, a1.v, false, bf1.v,
                                                     (short)0, acc, false, false);

        // Running argmin over this lane's 8 codes of the tile
        #pragma unroll
        for (int r = 0; r < 8; ++r) {
            float dct = acc[r];
            int code  = tile * 16 + half * 8 + r;
            if (dct < minval) { minval = dct; minidx = code; }
        }
    }

    // Combine the two lane-halves (both correspond to position lm): codes {.,0-7} vs {.,8-15}
    {
        float oval = __shfl_xor(minval, 16);
        int   oidx = __shfl_xor(minidx, 16);
        if (oval < minval || (oval == minval && oidx < minidx)) {
            minval = oval; minidx = oidx;
        }
    }

    const int pos = nbase + wave * 16 + lm;   // position n within batch image b

    // ---- Write index (lanes 0-15 only; 16 consecutive int32 -> coalesced) ----
    if (half == 0) {
        idx_out[(size_t)b * NN + pos] = minidx;
    }

    // ---- Gather embeddings in (B,D,H,W) layout: lane half selects d range ----
    // out[b, d, pos] = weight[minidx, d]; per fixed d, 16 lanes write consecutive n.
    const float* wsrc = weight + (size_t)minidx * DD + half * 32;
    float* dst = emb_out + (size_t)b * DD * NN + (size_t)(half * 32) * NN + pos;
    #pragma unroll
    for (int q = 0; q < 8; ++q) {
        float4 wv = *(const float4*)(wsrc + q * 4);
        dst[0] = wv.x; dst += NN;
        dst[0] = wv.y; dst += NN;
        dst[0] = wv.z; dst += NN;
        dst[0] = wv.w; dst += NN;
    }
}

extern "C" void kernel_launch(void* const* d_in, const int* in_sizes, int n_in,
                              void* d_out, int out_size, void* d_ws, size_t ws_size,
                              hipStream_t stream) {
    const float* enc    = (const float*)d_in[0];   // (B, D, H, W) f32
    const float* weight = (const float*)d_in[1];   // (K, D) f32

    // d_out = [indices (B*H*W int32) | embeddings (B*D*H*W f32)], concatenated flat
    int*   idx_out = (int*)d_out;
    float* emb_out = (float*)d_out + (size_t)BB * HH * WW;

    dim3 grid(BB * (NN / WGPOS));   // 1024 blocks
    dim3 block(NTHREADS);           // 8 wave32s
    hipLaunchKernelGGL(vq_argmin_gather_kernel, grid, block, 0, stream,
                       enc, weight, idx_out, emb_out);
}
